// DecoupledCls_63402307223787
// MI455X (gfx1250) — compile-verified, matrix operand order checked by reading, not run on hardware
//
#include <hip/hip_runtime.h>

// ---------------- problem constants ----------------
#define LL   2
#define BB   4
#define TT   2048
#define FF   1024
#define CC   200
#define HH   8
#define DFF  128
#define SS   (TT + CC)      // 2248
#define DH   (FF / HH)      // 128

typedef __attribute__((ext_vector_type(16))) __bf16 v16bf;
typedef __attribute__((ext_vector_type(8)))  __bf16 v8bf;
typedef __attribute__((ext_vector_type(8)))  float  v8f;

__device__ __forceinline__ unsigned short f2bf(float f) {
  unsigned u = __float_as_uint(f);
  unsigned r = 0x7fffu + ((u >> 16) & 1u);
  return (unsigned short)((u + r) >> 16);
}

__device__ __forceinline__ v8f wmma_bf16(v16bf a, v16bf b, v8f c) {
  return __builtin_amdgcn_wmma_f32_16x16x32_bf16(
      /*neg_a=*/false, a, /*neg_b=*/false, b,
      /*c_mod=*/(short)0, c, /*reuse_a=*/false, /*reuse_b=*/false);
}

// Build one 16-lane-row x 32-K fragment. `p` points at [row][k0] of a
// row-major bf16 tile; lanes 0-15 take K {0..7,16..23}, lanes 16-31 take
// K {8..15,24..31}: two contiguous 16B loads each.
__device__ __forceinline__ v16bf load_frag(const unsigned short* p, int half) {
  v8bf lo = *(const v8bf*)(p + half * 8);
  v8bf hi = *(const v8bf*)(p + 16 + half * 8);
  union { v16bf v; v8bf h[2]; } u;
  u.h[0] = lo; u.h[1] = hi;
  return u.v;
}

// CDNA5 async global->LDS copy (16B per lane), tracked by ASYNCcnt.
__device__ __forceinline__ void async_copy_b128(void* lds_ptr, const void* gptr) {
  unsigned laddr = (unsigned)(size_t)lds_ptr;   // low 32 bits = LDS byte offset
  asm volatile("global_load_async_to_lds_b128 %0, %1, off"
               :: "v"(laddr), "v"(gptr) : "memory");
}
__device__ __forceinline__ void async_wait0() {
  asm volatile("s_wait_asynccnt 0x0" ::: "memory");
}

// ---------------- small utility kernels ----------------
__global__ void cvt_bf16_kernel(const float* __restrict__ in,
                                unsigned short* __restrict__ out, long n) {
  long i = (long)blockIdx.x * blockDim.x + threadIdx.x;
  long stride = (long)gridDim.x * blockDim.x;
  for (; i < n; i += stride) out[i] = f2bf(in[i]);
}

// src (B,S,F): rows [0,T) = x, rows [T,S) = prototypes reshaped to (F,C).
__global__ void build_src_kernel(const float* __restrict__ x,
                                 const float* __restrict__ proto,
                                 float* __restrict__ srcf,
                                 unsigned short* __restrict__ srcb) {
  long n = (long)BB * SS * FF;
  long i = (long)blockIdx.x * blockDim.x + threadIdx.x;
  long stride = (long)gridDim.x * blockDim.x;
  for (; i < n; i += stride) {
    int f = (int)(i % FF);
    long rs = i / FF;
    int s = (int)(rs % SS);
    int b = (int)(rs / SS);
    float v;
    if (s < TT) v = x[((long)b * TT + s) * FF + f];
    else        v = proto[(long)f * CC + (s - TT)];
    srcf[i] = v;
    srcb[i] = f2bf(v);
  }
}

// emb_w (F,F,3) -> w_eff (F, 3F) with w_eff[o][dt*F+i] = emb_w[o][i][dt], bf16.
__global__ void repack_emb_kernel(const float* __restrict__ w,
                                  unsigned short* __restrict__ we) {
  long n = (long)FF * FF * 3;
  long i = (long)blockIdx.x * blockDim.x + threadIdx.x;
  long stride = (long)gridDim.x * blockDim.x;
  for (; i < n; i += stride) {
    int ii = (int)(i % FF);
    long od = i / FF;
    int dt = (int)(od % 3);
    int o  = (int)(od / 3);
    we[i] = f2bf(w[((long)o * FF + ii) * 3 + dt]);
  }
}

// featp (B, T+2, F) bf16, zero rows at both ends, from src (B,S,F) fp32.
__global__ void pad_feat_kernel(const float* __restrict__ src,
                                unsigned short* __restrict__ fp) {
  long n = (long)BB * (TT + 2) * FF;
  long i = (long)blockIdx.x * blockDim.x + threadIdx.x;
  long stride = (long)gridDim.x * blockDim.x;
  for (; i < n; i += stride) {
    int f = (int)(i % FF);
    long rb = i / FF;
    int tp = (int)(rb % (TT + 2));
    int b  = (int)(rb / (TT + 2));
    unsigned short v = 0;
    if (tp >= 1 && tp <= TT)
      v = f2bf(src[((long)b * SS + (tp - 1)) * FF + f]);
    fp[i] = v;
  }
}

// residual add + LayerNorm over F, writes fp32 + bf16 (may alias resid).
__global__ __launch_bounds__(256) void add_ln_kernel(
    const float* __restrict__ resid, const float* __restrict__ add,
    const float* __restrict__ gam, const float* __restrict__ bet,
    float* __restrict__ outf, unsigned short* __restrict__ outb) {
  long row = blockIdx.x;
  const float* r = resid + row * FF;
  const float* a = add + row * FF;
  int tid = threadIdx.x;
  __shared__ float red[256];
  float x[4];
  float s = 0.f;
#pragma unroll
  for (int i = 0; i < 4; ++i) {
    int c = tid + i * 256;
    x[i] = r[c] + a[c];
    s += x[i];
  }
  red[tid] = s; __syncthreads();
  for (int w = 128; w > 0; w >>= 1) { if (tid < w) red[tid] += red[tid + w]; __syncthreads(); }
  float mean = red[0] * (1.0f / FF);
  __syncthreads();
  float v = 0.f;
#pragma unroll
  for (int i = 0; i < 4; ++i) { float d = x[i] - mean; v += d * d; }
  red[tid] = v; __syncthreads();
  for (int w = 128; w > 0; w >>= 1) { if (tid < w) red[tid] += red[tid + w]; __syncthreads(); }
  float rstd = rsqrtf(red[0] * (1.0f / FF) + 1e-5f);
#pragma unroll
  for (int i = 0; i < 4; ++i) {
    int c = tid + i * 256;
    float y = (x[i] - mean) * rstd * gam[c] + bet[c];
    outf[row * FF + c] = y;
    outb[row * FF + c] = f2bf(y);
  }
}

// ---------------- bf16 WMMA GEMM: D = A(MxK) * W(NxK)^T + bias ----------------
// Block tile 128x128, 8 waves (4 along M, 2 along N), wave tile 32x64
// (2x4 wmma). Double-buffered LDS staged via async global->LDS copies.
#define BM 128
#define BN 128
__global__ __launch_bounds__(256) void gemm_bias_kernel(
    const unsigned short* __restrict__ A, long lda, long Abatch,
    const unsigned short* __restrict__ W,          // (N,K) row-major bf16
    const float* __restrict__ bias,                 // length N (or null)
    float* __restrict__ Df, long Dfrs, long Dfcs, long Dfbatch,
    unsigned short* __restrict__ Db, long Dbld, long Dbbatch,
    int M, int N, int K, int relu) {
  __shared__ __align__(16) unsigned short As[2][BM][32];
  __shared__ __align__(16) unsigned short Ws[2][BN][32];
  int bz = blockIdx.z;
  const unsigned short* Ab = A + (long)bz * Abatch;
  int m0 = blockIdx.x * BM, n0 = blockIdx.y * BN;
  int tid = threadIdx.x;
  int wave = tid >> 5, lane = tid & 31;
  int wm = wave >> 1, wn = wave & 1;
  int half = lane >> 4, l16 = lane & 15;
  v8f acc[2][4] = {};

  auto stage = [&](int buf, int k0) {
    // A tile: 128 rows x 32 K (16B chunks, 512 chunks, 2 per thread)
    for (int i = tid; i < BM * 4; i += 256) {
      int r = i >> 2, c = (i & 3) * 8;
      int gr = m0 + r;
      if (gr < M)
        async_copy_b128(&As[buf][r][c], Ab + (long)gr * lda + k0 + c);
      else
        *(uint4*)&As[buf][r][c] = make_uint4(0u, 0u, 0u, 0u);
    }
    // W tile: 128 rows x 32 K (always in range: N % 128 == 0)
    for (int i = tid; i < BN * 4; i += 256) {
      int r = i >> 2, c = (i & 3) * 8;
      async_copy_b128(&Ws[buf][r][c], W + (long)(n0 + r) * K + k0 + c);
    }
  };

  int nk = K >> 5;
  stage(0, 0);
  async_wait0();
  __syncthreads();
  for (int kb = 0; kb < nk; ++kb) {
    int cur = kb & 1;
    if (kb + 1 < nk) stage(1 - cur, (kb + 1) * 32);   // overlap with WMMAs
    v16bf afr[2], wfr[4];
#pragma unroll
    for (int sm = 0; sm < 2; ++sm)
      afr[sm] = load_frag(&As[cur][wm * 32 + sm * 16 + l16][0], half);
#pragma unroll
    for (int sn = 0; sn < 4; ++sn)
      wfr[sn] = load_frag(&Ws[cur][wn * 64 + sn * 16 + l16][0], half);
#pragma unroll
    for (int sm = 0; sm < 2; ++sm)
#pragma unroll
      for (int sn = 0; sn < 4; ++sn)
        acc[sm][sn] = wmma_bf16(afr[sm], wfr[sn], acc[sm][sn]);
    async_wait0();
    __syncthreads();
  }

  int colbase = n0 + wn * 64;
  int rowbase = m0 + wm * 32;
#pragma unroll
  for (int sm = 0; sm < 2; ++sm) {
#pragma unroll
    for (int sn = 0; sn < 4; ++sn) {
      int col = colbase + sn * 16 + l16;
      float bv = bias ? bias[col] : 0.f;
      v8f v = acc[sm][sn];
#pragma unroll
      for (int g = 0; g < 8; ++g) {
        int row = rowbase + sm * 16 + g + half * 8;
        if (row < M) {
          float y = v[g] + bv;
          if (relu) y = fmaxf(y, 0.f);
          if (Df) Df[(long)bz * Dfbatch + (long)row * Dfrs + (long)col * Dfcs] = y;
          if (Db) Db[(long)bz * Dbbatch + (long)row * Dbld + col] = f2bf(y);
        }
      }
    }
  }
}

// ---------------- flash attention (bf16 WMMA, online softmax) ----------------
// grid: (ceil(S/128), H, B); 8 waves, each owns 16 query rows.
__global__ __launch_bounds__(256) void flash_attn_kernel(
    const unsigned short* __restrict__ Qb, const unsigned short* __restrict__ Kb,
    const unsigned short* __restrict__ Vb, unsigned short* __restrict__ Ob,
    float scale) {
  __shared__ __align__(16) unsigned short Kt[32][DH];       // keys row-major
  __shared__ __align__(16) unsigned short Vt[DH][32];       // V transposed
  __shared__ __align__(16) unsigned short Pt[8][16][32];    // per-wave P tiles
  int b = blockIdx.z, h = blockIdx.y;
  int qb0 = blockIdx.x * 128;
  int tid = threadIdx.x, wave = tid >> 5, lane = tid & 31;
  int half = lane >> 4, l16 = lane & 15;
  long base = ((long)b * SS) * FF + (long)h * DH;

  int qrow = qb0 + wave * 16 + l16;
  int qcl = qrow < SS ? qrow : SS - 1;
  v16bf aQ[4];
#pragma unroll
  for (int c = 0; c < 4; ++c)
    aQ[c] = load_frag(Qb + base + (long)qcl * FF + c * 32, half);

  v8f O[8] = {};
  float mrow[8], lrow[8];
#pragma unroll
  for (int g = 0; g < 8; ++g) { mrow[g] = -1e30f; lrow[g] = 0.f; }

  int nk = (SS + 31) / 32;
  for (int kb = 0; kb < nk; ++kb) {
    int k0 = kb * 32;
    __syncthreads();
    // stage K rows (32 x 128) via async global->LDS; prefetch next tile
    for (int i = tid; i < 512; i += 256) {
      int r = i >> 4, c = (i & 15) * 8;
      int kr = k0 + r;
      if (kr < SS) {
        async_copy_b128(&Kt[r][c], Kb + base + (long)kr * FF + c);
        if (kr + 32 < SS)
          __builtin_prefetch(Kb + base + (long)(kr + 32) * FF + c, 0, 0);
      } else {
        *(uint4*)&Kt[r][c] = make_uint4(0u, 0u, 0u, 0u);
      }
    }
    // stage V transposed (128 x 32)
    for (int i = tid; i < 32 * DH; i += 256) {
      int kk = i >> 7, d = i & 127;
      int kr = k0 + kk;
      unsigned short v = 0;
      if (kr < SS) v = Vb[base + (long)kr * FF + d];
      Vt[d][kk] = v;
    }
    async_wait0();
    __syncthreads();

    // scores S = Q * K^T (two 16-key column tiles)
    v8f sc[2];
#pragma unroll
    for (int j = 0; j < 2; ++j) {
      v8f acc = {};
#pragma unroll
      for (int c = 0; c < 4; ++c) {
        v16bf bK = load_frag(&Kt[j * 16 + l16][c * 32], half);
        acc = wmma_bf16(aQ[c], bK, acc);
      }
      int kcol = k0 + j * 16 + l16;
      float msk = (kcol < SS) ? 0.f : -1e30f;
#pragma unroll
      for (int g = 0; g < 8; ++g) acc[g] = acc[g] * scale + msk;
      sc[j] = acc;
    }

    // row max across the 16 lanes of each half, combine tiles
    float fcorr[8];
#pragma unroll
    for (int g = 0; g < 8; ++g) {
      float v = fmaxf(sc[0][g], sc[1][g]);
#pragma unroll
      for (int off = 8; off >= 1; off >>= 1)
        v = fmaxf(v, __shfl_xor(v, off, 32));
      float mn = fmaxf(mrow[g], v);
      fcorr[g] = __expf(mrow[g] - mn);
      mrow[g] = mn;
    }
    // P = exp(S - m), write to per-wave LDS tile (C-layout -> A-layout)
#pragma unroll
    for (int j = 0; j < 2; ++j) {
      v8f p;
#pragma unroll
      for (int g = 0; g < 8; ++g) p[g] = __expf(sc[j][g] - mrow[g]);
      sc[j] = p;
#pragma unroll
      for (int g = 0; g < 8; ++g)
        Pt[wave][g + half * 8][j * 16 + l16] = f2bf(p[g]);
    }
    // row sums + rescale accumulators
#pragma unroll
    for (int g = 0; g < 8; ++g) {
      float v = sc[0][g] + sc[1][g];
#pragma unroll
      for (int off = 8; off >= 1; off >>= 1) v += __shfl_xor(v, off, 32);
      lrow[g] = lrow[g] * fcorr[g] + v;
    }
#pragma unroll
    for (int ct = 0; ct < 8; ++ct)
#pragma unroll
      for (int g = 0; g < 8; ++g) O[ct][g] *= fcorr[g];

    // O += P * V
    v16bf aP = load_frag(&Pt[wave][l16][0], half);
#pragma unroll
    for (int ct = 0; ct < 8; ++ct) {
      v16bf bV = load_frag(&Vt[ct * 16 + l16][0], half);
      O[ct] = wmma_bf16(aP, bV, O[ct]);
    }
  }

  // finalize: divide by l, store bf16
#pragma unroll
  for (int g = 0; g < 8; ++g) {
    int r = qb0 + wave * 16 + g + half * 8;
    if (r >= SS) continue;
    float inv = 1.f / fmaxf(lrow[g], 1e-20f);
#pragma unroll
    for (int ct = 0; ct < 8; ++ct)
      Ob[base + (long)r * FF + ct * 16 + l16] = f2bf(O[ct][g] * inv);
  }
}

// ---------------- host side ----------------
static inline void launch_gemm(hipStream_t s,
    const unsigned short* A, long lda, long Abatch,
    const unsigned short* W, const float* bias,
    float* Df, long rs, long cs, long Dfb,
    unsigned short* Db, long Dbld, long Dbb,
    int M, int N, int K, int relu, int batches) {
  dim3 g((M + BM - 1) / BM, N / BN, batches);
  gemm_bias_kernel<<<g, dim3(256), 0, s>>>(A, lda, Abatch, W, bias,
                                           Df, rs, cs, Dfb, Db, Dbld, Dbb,
                                           M, N, K, relu);
}

extern "C" void kernel_launch(void* const* d_in, const int* in_sizes, int n_in,
                              void* d_out, int out_size, void* d_ws, size_t ws_size,
                              hipStream_t stream) {
  const float* x     = (const float*)d_in[0];
  const float* proto = (const float*)d_in[1];
  const float* wq    = (const float*)d_in[2];
  const float* bq    = (const float*)d_in[3];
  const float* wk    = (const float*)d_in[4];
  const float* bk    = (const float*)d_in[5];
  const float* in_w  = (const float*)d_in[6];
  const float* in_b  = (const float*)d_in[7];
  const float* out_w = (const float*)d_in[8];
  const float* out_b = (const float*)d_in[9];
  const float* l1_w  = (const float*)d_in[10];
  const float* l1_b  = (const float*)d_in[11];
  const float* l2_w  = (const float*)d_in[12];
  const float* l2_b  = (const float*)d_in[13];
  const float* ln1_g = (const float*)d_in[14];
  const float* ln1_b = (const float*)d_in[15];
  const float* ln2_g = (const float*)d_in[16];
  const float* ln2_b = (const float*)d_in[17];
  const float* emb_w = (const float*)d_in[18];
  const float* emb_b = (const float*)d_in[19];
  float* out = (float*)d_out;

  const long NS = (long)BB * SS;       // token rows
  const long NSF = NS * FF;

  char* ws = (char*)d_ws;
  size_t off = 0;
  auto alloc = [&](size_t bytes) -> void* {
    void* p = ws + off;
    off = (off + bytes + 255) & ~(size_t)255;
    return p;
  };
  float*          srcf = (float*)alloc(NSF * 4);
  unsigned short* srcb = (unsigned short*)alloc(NSF * 2);
  unsigned short* qb   = (unsigned short*)alloc(NSF * 2);
  unsigned short* kb   = (unsigned short*)alloc(NSF * 2);
  unsigned short* Qp   = (unsigned short*)alloc(NSF * 2);
  unsigned short* Kp   = (unsigned short*)alloc(NSF * 2);
  unsigned short* Vp   = (unsigned short*)alloc(NSF * 2);
  unsigned short* Obuf = (unsigned short*)alloc(NSF * 2);
  float*          af   = (float*)alloc(NSF * 4);           // attn-out / ffn-out fp32
  unsigned short* h1b  = (unsigned short*)alloc(NS * DFF * 2);
  unsigned short* wqb  = (unsigned short*)alloc((size_t)LL * FF * FF * 2);
  unsigned short* wkb  = (unsigned short*)alloc((size_t)LL * FF * FF * 2);
  unsigned short* inwb = (unsigned short*)alloc((size_t)LL * 3 * FF * FF * 2);
  unsigned short* outwb= (unsigned short*)alloc((size_t)LL * FF * FF * 2);
  unsigned short* l1wb = (unsigned short*)alloc((size_t)LL * DFF * FF * 2);
  unsigned short* l2wb = (unsigned short*)alloc((size_t)LL * FF * DFF * 2);
  unsigned short* embb = (unsigned short*)alloc((size_t)FF * 3 * FF * 2);
  unsigned short* featp= (unsigned short*)alloc((size_t)BB * (TT + 2) * FF * 2);

  auto cvt = [&](const float* src, unsigned short* dst, long n) {
    int blocks = (int)((n + 255) / 256);
    if (blocks > 16384) blocks = 16384;
    cvt_bf16_kernel<<<blocks, 256, 0, stream>>>(src, dst, n);
  };
  cvt(wq,    wqb,  (long)LL * FF * FF);
  cvt(wk,    wkb,  (long)LL * FF * FF);
  cvt(in_w,  inwb, (long)LL * 3 * FF * FF);
  cvt(out_w, outwb,(long)LL * FF * FF);
  cvt(l1_w,  l1wb, (long)LL * DFF * FF);
  cvt(l2_w,  l2wb, (long)LL * FF * DFF);
  repack_emb_kernel<<<4096, 256, 0, stream>>>(emb_w, embb);
  build_src_kernel<<<16384, 256, 0, stream>>>(x, proto, srcf, srcb);

  const int M = (int)NS;
  const float scale = 0.08838834764831845f;  // 1/sqrt(128)

  for (int l = 0; l < LL; ++l) {
    const unsigned short* Wq = wqb  + (long)l * FF * FF;
    const unsigned short* Wk = wkb  + (long)l * FF * FF;
    const unsigned short* Wi = inwb + (long)l * 3 * FF * FF;
    const unsigned short* Wo = outwb+ (long)l * FF * FF;
    const unsigned short* W1 = l1wb + (long)l * DFF * FF;
    const unsigned short* W2 = l2wb + (long)l * FF * DFF;

    // q = src Wq^T + bq ; k = src Wk^T + bk  (bf16 out)
    launch_gemm(stream, srcb, FF, 0, Wq, bq + (long)l * FF,
                nullptr, 0, 0, 0, qb, FF, 0, M, FF, FF, 0, 1);
    launch_gemm(stream, srcb, FF, 0, Wk, bk + (long)l * FF,
                nullptr, 0, 0, 0, kb, FF, 0, M, FF, FF, 0, 1);
    // packed in-proj: Q = q Win_q^T, K = k Win_k^T, V = src Win_v^T
    launch_gemm(stream, qb, FF, 0, Wi, in_b + (long)l * 3 * FF,
                nullptr, 0, 0, 0, Qp, FF, 0, M, FF, FF, 0, 1);
    launch_gemm(stream, kb, FF, 0, Wi + (long)FF * FF, in_b + (long)l * 3 * FF + FF,
                nullptr, 0, 0, 0, Kp, FF, 0, M, FF, FF, 0, 1);
    launch_gemm(stream, srcb, FF, 0, Wi + 2L * FF * FF, in_b + (long)l * 3 * FF + 2 * FF,
                nullptr, 0, 0, 0, Vp, FF, 0, M, FF, FF, 0, 1);
    // attention
    dim3 ag((SS + 127) / 128, HH, BB);
    flash_attn_kernel<<<ag, dim3(256), 0, stream>>>(Qp, Kp, Vp, Obuf, scale);
    // out-proj (fp32 out for residual)
    launch_gemm(stream, Obuf, FF, 0, Wo, out_b + (long)l * FF,
                af, FF, 1, 0, nullptr, 0, 0, M, FF, FF, 0, 1);
    // src = LN(src + attn)
    add_ln_kernel<<<(int)NS, 256, 0, stream>>>(srcf, af,
        ln1_g + (long)l * FF, ln1_b + (long)l * FF, srcf, srcb);
    // FFN
    launch_gemm(stream, srcb, FF, 0, W1, l1_b + (long)l * DFF,
                nullptr, 0, 0, 0, h1b, DFF, 0, M, DFF, FF, 1, 1);
    launch_gemm(stream, h1b, DFF, 0, W2, l2_b + (long)l * FF,
                af, FF, 1, 0, nullptr, 0, 0, M, FF, DFF, 0, 1);
    // src = LN(src + ffn)
    add_ln_kernel<<<(int)NS, 256, 0, stream>>>(srcf, af,
        ln2_g + (long)l * FF, ln2_b + (long)l * FF, srcf, srcb);
  }

  // Conv1d(k=3,pad=1)+ReLU as GEMM: K = 3F over padded contiguous rows,
  // output written directly in (B, F, T) order via strided fp32 epilogue.
  pad_feat_kernel<<<16384, 256, 0, stream>>>(srcf, featp);
  launch_gemm(stream, featp, FF, (long)(TT + 2) * FF, embb, emb_b,
              out, /*row(t) stride*/ 1, /*col(o) stride*/ TT, (long)FF * TT,
              nullptr, 0, 0, TT, FF, 3 * FF, /*relu=*/1, /*batches=*/BB);
}
